// SDPA_55937654063667
// MI455X (gfx1250) — compile-verified, matrix operand order checked by reading, not run on hardware
//
#include <hip/hip_runtime.h>

typedef __attribute__((ext_vector_type(16))) _Float16     v16h;
typedef __attribute__((ext_vector_type(8)))  float        v8f;
typedef __attribute__((ext_vector_type(4)))  float        f32x4;
typedef __attribute__((ext_vector_type(4)))  unsigned int u32x4;

#define B_    2
#define H_    8
#define S_    2048
#define D_    64
#define KB    64            // keys per block iteration
#define WAVES 8
#define QW    16            // queries per wave
#define QB    (WAVES * QW)  // 128 queries per workgroup

// log2(e) / sqrt(D): folds softmax scale AND base-2 exp into Q once.
#define QSCALE 0.18033688011112042f

// DPP lane-xor within 16-lane rows (ROW_XMASK = 0x160 | mask): pure-VALU butterfly.
template<int CTRL>
__device__ __forceinline__ float dpp_f32(float x) {
    int i = __builtin_bit_cast(int, x);
    i = __builtin_amdgcn_update_dpp(i, i, CTRL, 0xF, 0xF, true);
    return __builtin_bit_cast(float, i);
}
__device__ __forceinline__ float rowmax16(float x) {
    x = fmaxf(x, dpp_f32<0x161>(x));   // xor 1
    x = fmaxf(x, dpp_f32<0x162>(x));   // xor 2
    x = fmaxf(x, dpp_f32<0x164>(x));   // xor 4
    x = fmaxf(x, dpp_f32<0x168>(x));   // xor 8
    return x;
}
__device__ __forceinline__ float rowsum16(float x) {
    x += dpp_f32<0x161>(x);
    x += dpp_f32<0x162>(x);
    x += dpp_f32<0x164>(x);
    x += dpp_f32<0x168>(x);
    return x;
}

__launch_bounds__(256, 1)
__global__ void sdpa_fa_wmma(const float* __restrict__ Q,
                             const float* __restrict__ K,
                             const float* __restrict__ V,
                             float* __restrict__ O)
{
    // Double-buffered K/V tiles + per-wave P staging: 48 KB total.
    __shared__ __align__(16) _Float16 Ks[2][KB][D_];     // 2 x 8 KB, [key][d]
    __shared__ __align__(16) _Float16 Vt[2][D_][KB];     // 2 x 8 KB, [d][key]
    __shared__ __align__(16) _Float16 Ps[WAVES][16][KB]; // 16 KB

    const int tid   = threadIdx.x;
    const int w     = tid >> 5;
    const int lane  = tid & 31;
    const int lh    = lane & 15;       // lane within 16-lane half
    const int hi    = lane >> 4;       // which half
    const int kbase = hi ? 8 : 0;      // fragment contraction base for this lane

    const int nqb = S_ / QB;
    const int bid = blockIdx.x;
    const int bh  = bid / nqb;         // fused (b*H + h)
    const int qb  = bid % nqb;
    const int b   = bh / H_;
    const int h   = bh % H_;

    const float* Qp = Q + (size_t)bh * S_ * D_;
    const float* Kp = K + (size_t)bh * S_ * D_;
    const float* Vp = V + (size_t)bh * S_ * D_;

    const int q0 = qb * QB + w * QW;

    // ---- Q A-fragments, pre-scaled by log2(e)/sqrt(D) (f32 -> f16) ----
    v16h aq[2];
    {
        const float* qrow = Qp + (size_t)(q0 + lh) * D_;
        #pragma unroll
        for (int dc = 0; dc < 2; ++dc) {
            const float* p0 = qrow + dc * 32 + kbase;
            const float* p1 = qrow + dc * 32 + 16 + kbase;
            #pragma unroll
            for (int j = 0; j < 8; ++j) {
                aq[dc][j]     = (_Float16)(p0[j] * QSCALE);
                aq[dc][j + 8] = (_Float16)(p1[j] * QSCALE);
            }
        }
    }

    v8f zf = {};
    v8f acc[4];
    #pragma unroll
    for (int nt = 0; nt < 4; ++nt) acc[nt] = zf;
    float mrow[8], lrow[8];            // lrow: per-lane PARTIAL sums (log2 domain)
    #pragma unroll
    for (int r = 0; r < 8; ++r) { mrow[r] = -3.0e38f; lrow[r] = 0.0f; }

    const int keyld = tid >> 2;          // 0..63: staged key row
    const int dseg  = (tid & 3) * 16;    // 0,16,32,48: 16-wide d segment

    // ---- pipeline preload: block 0 K/V into registers ----
    f32x4 kr[4], vr[4];
    {
        const f32x4* krow = (const f32x4*)(Kp + (size_t)keyld * D_ + dseg);
        const f32x4* vrow = (const f32x4*)(Vp + (size_t)keyld * D_ + dseg);
        #pragma unroll
        for (int j = 0; j < 4; ++j) { kr[j] = krow[j]; vr[j] = vrow[j]; }
    }

    for (int kb = 0; kb < S_; kb += KB) {
        const int buf = (kb / KB) & 1;

        // ---- drain staged registers into LDS (f32 -> f16, V transposed) ----
        {
            union { _Float16 h8[8]; u32x4 u; } pk0, pk1;
            #pragma unroll
            for (int j = 0; j < 8; ++j) {
                pk0.h8[j] = (_Float16)kr[j >> 2][j & 3];
                pk1.h8[j] = (_Float16)kr[(j + 8) >> 2][j & 3];
            }
            *(u32x4*)&Ks[buf][keyld][dseg]     = pk0.u;
            *(u32x4*)&Ks[buf][keyld][dseg + 8] = pk1.u;
            #pragma unroll
            for (int j = 0; j < 16; ++j)
                Vt[buf][dseg + j][keyld] = (_Float16)vr[j >> 2][j & 3];
        }
        __syncthreads();

        // ---- issue next block's global loads; consumed next iteration ----
        if (kb + KB < S_) {  // uniform condition
            const f32x4* krow = (const f32x4*)(Kp + (size_t)(kb + KB + keyld) * D_ + dseg);
            const f32x4* vrow = (const f32x4*)(Vp + (size_t)(kb + KB + keyld) * D_ + dseg);
            #pragma unroll
            for (int j = 0; j < 4; ++j) { kr[j] = krow[j]; vr[j] = vrow[j]; }
        }

        // ---- scores: S[16q x 64k] in four 16x16 fragments, 8 WMMAs ----
        v8f c[4];
        #pragma unroll
        for (int kk = 0; kk < 4; ++kk) c[kk] = zf;
        #pragma unroll
        for (int dc = 0; dc < 2; ++dc) {
            #pragma unroll
            for (int kk = 0; kk < 4; ++kk) {
                union { v16h v; u32x4 u[2]; } bk;
                bk.u[0] = *(const u32x4*)&Ks[buf][kk * 16 + lh][dc * 32 + kbase];
                bk.u[1] = *(const u32x4*)&Ks[buf][kk * 16 + lh][dc * 32 + 16 + kbase];
                c[kk] = __builtin_amdgcn_wmma_f32_16x16x32_f16(false, aq[dc], false, bk.v,
                                                               (short)0, c[kk], false, false);
            }
        }

        // ---- online softmax, base-2 domain, DPP-only reductions ----
        float p[4][8], alpha[8];
        #pragma unroll
        for (int r = 0; r < 8; ++r) {
            float mx = fmaxf(fmaxf(c[0][r], c[1][r]), fmaxf(c[2][r], c[3][r]));
            mx = rowmax16(mx);
            float mn = fmaxf(mrow[r], mx);
            alpha[r] = __builtin_amdgcn_exp2f(mrow[r] - mn);
            mrow[r]  = mn;
            float ls = 0.0f;
            #pragma unroll
            for (int f = 0; f < 4; ++f) {
                p[f][r] = __builtin_amdgcn_exp2f(c[f][r] - mn);
                ls += p[f][r];
            }
            lrow[r] = lrow[r] * alpha[r] + ls;   // lane-local partial; reduce at end
        }
        #pragma unroll
        for (int nt = 0; nt < 4; ++nt)
            #pragma unroll
            for (int r = 0; r < 8; ++r)
                acc[nt][r] *= alpha[r];

        // ---- P: C-layout -> A-layout via per-wave LDS tile ----
        #pragma unroll
        for (int r = 0; r < 8; ++r) {
            const int row = r + 8 * hi;
            #pragma unroll
            for (int f = 0; f < 4; ++f)
                Ps[w][row][f * 16 + lh] = (_Float16)p[f][r];
        }
        asm volatile("s_wait_dscnt 0" ::: "memory");  // same-wave DS order fence

        union { v16h v; u32x4 u[2]; } ap[2];
        #pragma unroll
        for (int kc = 0; kc < 2; ++kc) {
            ap[kc].u[0] = *(const u32x4*)&Ps[w][lh][kc * 32 + kbase];
            ap[kc].u[1] = *(const u32x4*)&Ps[w][lh][kc * 32 + 16 + kbase];
        }

        // ---- O += P(16x64) * V(64x64): 4 N-tiles x 2 key-chunks, 8 WMMAs ----
        #pragma unroll
        for (int nt = 0; nt < 4; ++nt) {
            #pragma unroll
            for (int kc = 0; kc < 2; ++kc) {
                union { v16h v; u32x4 u[2]; } bv;
                bv.u[0] = *(const u32x4*)&Vt[buf][nt * 16 + lh][kc * 32 + kbase];
                bv.u[1] = *(const u32x4*)&Vt[buf][nt * 16 + lh][kc * 32 + 16 + kbase];
                acc[nt] = __builtin_amdgcn_wmma_f32_16x16x32_f16(false, ap[kc].v, false, bv.v,
                                                                 (short)0, acc[nt], false, false);
            }
        }
    }

    // ---- epilogue: reduce l across lanes, normalize, store O[B, S, H, D] ----
    #pragma unroll
    for (int r = 0; r < 8; ++r) {
        const float lsum = rowsum16(lrow[r]);
        const float inv  = 1.0f / lsum;
        const int   qrow = q0 + r + 8 * hi;
        float* orow = O + ((size_t)(b * S_ + qrow) * H_ + h) * D_;
        #pragma unroll
        for (int nt = 0; nt < 4; ++nt)
            orow[nt * 16 + lh] = acc[nt][r] * inv;
    }
}

extern "C" void kernel_launch(void* const* d_in, const int* in_sizes, int n_in,
                              void* d_out, int out_size, void* d_ws, size_t ws_size,
                              hipStream_t stream) {
    const float* q = (const float*)d_in[0];
    const float* k = (const float*)d_in[1];
    const float* v = (const float*)d_in[2];
    float* o = (float*)d_out;
    (void)in_sizes; (void)n_in; (void)out_size; (void)d_ws; (void)ws_size;

    dim3 grid(B_ * H_ * (S_ / QB));  // 256 workgroups
    dim3 block(256);                 // 8 waves (wave32)
    sdpa_fa_wmma<<<grid, block, 0, stream>>>(q, k, v, o);
}
